// MixtureOfExperts_18090402251417
// MI455X (gfx1250) — matrix operand order repaired and checked by reading, not compiled, not run on hardware
//
#include <hip/hip_runtime.h>
#include <hip/hip_bf16.h>
#include <stdint.h>

#define NUM_EXPERTS 8
#define HIDDEN 1024
#define INTER 4096
#define TOKENS 128

typedef __bf16 bf16_t;
typedef __attribute__((ext_vector_type(16))) __bf16 v16bf;
typedef __attribute__((ext_vector_type(8)))  __bf16 v8bf;
typedef __attribute__((ext_vector_type(4)))  __bf16 v4bf;
typedef __attribute__((ext_vector_type(8)))  float  v8f;
typedef __attribute__((ext_vector_type(4)))  float  v4f;

// GCC-style int4, matching the async builtin's parameter type exactly
typedef int v4i_g __attribute__((vector_size(16)));
typedef __attribute__((address_space(1))) v4i_g GV4i;
typedef __attribute__((address_space(3))) v4i_g LV4i;
typedef __attribute__((address_space(3))) void  LASVoid;

#define LDS_K 64
#define LDS_STRIDE 72   // 64 bf16 + 8 pad -> 144B rows (16B aligned)
#define ASYNC_CHUNKS 4  // b128 chunks per thread per A-tile stage

// ---- CDNA5 async global->LDS copy (ASYNCcnt-tracked), 16B per lane ----
static __device__ __forceinline__ void async_copy_b128(const bf16_t* g, bf16_t* l) {
#if __has_builtin(__builtin_amdgcn_global_load_async_to_lds_b128)
  __builtin_amdgcn_global_load_async_to_lds_b128(
      (GV4i*)(bf16_t*)g, (LV4i*)l, 0, 0);
#else
  unsigned lds_off = (unsigned)(uintptr_t)(LASVoid*)l;
  asm volatile("global_load_async_to_lds_b128 %0, %1, off"
               :: "v"(lds_off), "v"(g) : "memory");
#endif
}

template <int N>
static __device__ __forceinline__ void wait_asynccnt() {
#if __has_builtin(__builtin_amdgcn_s_wait_asynccnt)
  __builtin_amdgcn_s_wait_asynccnt(N);
#else
  asm volatile("s_wait_asynccnt %0" :: "i"(N) : "memory");
#endif
}

// ---- fragment loaders (per CDNA5 WMMA VGPR layout tables) ----
// A 16x32 bf16: lane l(<16): row l, K = kb..kb+7 and kb+16..kb+23; kb=(lane>=16)?8:0.
static __device__ __forceinline__ v16bf load_frag_a(const bf16_t* lds, int row, int kbase) {
  const bf16_t* p = lds + row * LDS_STRIDE + kbase;
  v8bf lo = *(const v8bf*)(p);
  v8bf hi = *(const v8bf*)(p + 16);
  return __builtin_shufflevector(lo, hi, 0,1,2,3,4,5,6,7,8,9,10,11,12,13,14,15);
}
// B 32x16 bf16: lane holds column lane%16, K = koff..koff+7 and koff+16..koff+23; koff=(lane>=16)?8:0.
static __device__ __forceinline__ v16bf load_frag_b(const bf16_t* lds, int col, int koff) {
  const bf16_t* p = lds + col * LDS_STRIDE + koff;
  v8bf lo = *(const v8bf*)(p);
  v8bf hi = *(const v8bf*)(p + 16);
  return __builtin_shufflevector(lo, hi, 0,1,2,3,4,5,6,7,8,9,10,11,12,13,14,15);
}

static __device__ __forceinline__ v8f wmma_bf16(v16bf a, v16bf b, v8f c) {
  return __builtin_amdgcn_wmma_f32_16x16x32_bf16(false, a, false, b, (short)0, c, false, false);
}

static __device__ __forceinline__ v4bf cvt4(v4f v) {
  return (v4bf){(bf16_t)v.x, (bf16_t)v.y, (bf16_t)v.z, (bf16_t)v.w};
}

// ---------------- Stage 0: X fp32 -> bf16 (one-time, 256 KB) ----------------
__global__ __launch_bounds__(256) void moe_cvt_x_kernel(
    const float* __restrict__ X, bf16_t* __restrict__ Xb) {
  const int i = (blockIdx.x * 256 + threadIdx.x) * 4;   // 128 blocks cover 131072 elems
  v4f v = *(const v4f*)(X + i);
  *(v4bf*)(Xb + i) = cvt4(v);
}

// ---------------- Stage 1: routing (softmax + top-2 + renorm) ----------------
__global__ __launch_bounds__(TOKENS) void moe_routing_kernel(
    const float* __restrict__ routing, float* __restrict__ coef) {
  const int t = threadIdx.x;
  float r[NUM_EXPERTS];
  float mx = -3.4e38f;
  #pragma unroll
  for (int e = 0; e < NUM_EXPERTS; ++e) { r[e] = routing[t * NUM_EXPERTS + e]; mx = fmaxf(mx, r[e]); }
  #pragma unroll
  for (int e = 0; e < NUM_EXPERTS; ++e) r[e] = __expf(r[e] - mx);  // softmax denom cancels in renorm
  int i1 = 0; float v1 = r[0];
  #pragma unroll
  for (int e = 1; e < NUM_EXPERTS; ++e) if (r[e] > v1) { v1 = r[e]; i1 = e; }
  int i2 = -1; float v2 = -1.0f;
  #pragma unroll
  for (int e = 0; e < NUM_EXPERTS; ++e) if (e != i1 && r[e] > v2) { v2 = r[e]; i2 = e; }
  const float inv = 1.0f / (v1 + v2);
  #pragma unroll
  for (int e = 0; e < NUM_EXPERTS; ++e) coef[t * NUM_EXPERTS + e] = 0.0f;
  coef[t * NUM_EXPERTS + i1] = v1 * inv;
  coef[t * NUM_EXPERTS + i2] = v2 * inv;
}

// ---------------- Stage 2: fc1 (both halves) + swiglu -> act (bf16) ----------------
// grid: (INTER/64, NUM_EXPERTS), block 256 (8 waves: 2 in M x 4 in N)
// Per WG: 128(M) x 64(N) of linear half AND gate half; K staged 64-wide.
// X tile: async-copied bf16, ping-pong double buffered. W tiles: reg-staged double buffer.
__global__ __launch_bounds__(256) void moe_fc1_kernel(
    const bf16_t* __restrict__ Xb, const float* __restrict__ W1, bf16_t* __restrict__ act) {
  __shared__ bf16_t ldsX0[TOKENS * LDS_STRIDE];
  __shared__ bf16_t ldsX1[TOKENS * LDS_STRIDE];
  __shared__ bf16_t ldsWL[64 * LDS_STRIDE];
  __shared__ bf16_t ldsWG[64 * LDS_STRIDE];

  const int n0   = blockIdx.x * 64;
  const int e    = blockIdx.y;
  const int tid  = threadIdx.x;
  const int lane = tid & 31;
  const int wave = tid >> 5;
  const int wm   = wave & 1;    // M base = wm*64
  const int wn   = wave >> 1;   // N base = wn*16

  const float* w1e = W1 + (size_t)e * (2 * INTER) * HIDDEN;

  v8f accL[4] = {};
  v8f accG[4] = {};
  v4f wl[4], wg[4];

  // ---- preload stage 0 ----
  #pragma unroll
  for (int i = 0; i < ASYNC_CHUNKS; ++i) {
    const int idx = tid + i * 256;          // 0..1023
    const int row = idx >> 3;               // 8 chunks per 64-elem row
    const int c8  = (idx & 7) << 3;
    async_copy_b128(Xb + (size_t)row * HIDDEN + c8, ldsX0 + row * LDS_STRIDE + c8);
  }
  #pragma unroll
  for (int i = 0; i < 4; ++i) {
    const int idx4 = tid + i * 256;         // 1024 quads per 64x64 fp32 tile
    const int row  = idx4 >> 4;
    const int c4   = (idx4 & 15) << 2;
    wl[i] = __builtin_nontemporal_load((const v4f*)(w1e + (size_t)(n0 + row) * HIDDEN + c4));
    wg[i] = __builtin_nontemporal_load((const v4f*)(w1e + (size_t)(INTER + n0 + row) * HIDDEN + c4));
  }

  const int S = HIDDEN / LDS_K;             // 16 stages
  for (int s = 0; s < S; ++s) {
    const bf16_t* ldsXc = (s & 1) ? ldsX1 : ldsX0;
    bf16_t*       ldsXn = (s & 1) ? ldsX0 : ldsX1;

    __syncthreads();   // previous stage's compute done -> safe to overwrite W tiles / ldsXn

    // commit staged W regs for stage s
    #pragma unroll
    for (int i = 0; i < 4; ++i) {
      const int idx4 = tid + i * 256;
      const int row  = idx4 >> 4;
      const int c4   = (idx4 & 15) << 2;
      *(v4bf*)(ldsWL + row * LDS_STRIDE + c4) = cvt4(wl[i]);
      *(v4bf*)(ldsWG + row * LDS_STRIDE + c4) = cvt4(wg[i]);
    }

    if (s + 1 < S) {
      const int k1 = (s + 1) * LDS_K;
      #pragma unroll
      for (int i = 0; i < ASYNC_CHUNKS; ++i) {
        const int idx = tid + i * 256;
        const int row = idx >> 3;
        const int c8  = (idx & 7) << 3;
        async_copy_b128(Xb + (size_t)row * HIDDEN + k1 + c8, ldsXn + row * LDS_STRIDE + c8);
      }
      #pragma unroll
      for (int i = 0; i < 4; ++i) {
        const int idx4 = tid + i * 256;
        const int row  = idx4 >> 4;
        const int c4   = (idx4 & 15) << 2;
        wl[i] = __builtin_nontemporal_load((const v4f*)(w1e + (size_t)(n0 + row) * HIDDEN + k1 + c4));
        wg[i] = __builtin_nontemporal_load((const v4f*)(w1e + (size_t)(INTER + n0 + row) * HIDDEN + k1 + c4));
      }
      wait_asynccnt<ASYNC_CHUNKS>();  // only the just-issued prefetch may remain -> stage-s X landed
    } else {
      wait_asynccnt<0>();
    }
    __syncthreads();

    #pragma unroll
    for (int kk = 0; kk < LDS_K; kk += 32) {
      const int arow = wm * 64 + (lane & 15);
      const int kbA  = kk + ((lane >> 4) << 3);
      v16bf a[4];
      #pragma unroll
      for (int mi = 0; mi < 4; ++mi) a[mi] = load_frag_a(ldsXc, arow + mi * 16, kbA);
      const int bcol = wn * 16 + (lane & 15);
      const int kbB  = kk + ((lane >> 4) << 3);
      v16bf bl = load_frag_b(ldsWL, bcol, kbB);
      v16bf bg = load_frag_b(ldsWG, bcol, kbB);
      #pragma unroll
      for (int mi = 0; mi < 4; ++mi) {
        accL[mi] = wmma_bf16(a[mi], bl, accL[mi]);
        accG[mi] = wmma_bf16(a[mi], bg, accG[mi]);
      }
    }
  }

  // swiglu epilogue: act = silu(lin) * gate, store bf16
  bf16_t* actE = act + (size_t)e * TOKENS * INTER;
  const int n = n0 + wn * 16 + (lane & 15);
  #pragma unroll
  for (int mi = 0; mi < 4; ++mi) {
    const int mbase = wm * 64 + mi * 16 + ((lane >> 4) << 3);
    #pragma unroll
    for (int i = 0; i < 8; ++i) {
      const float xl = accL[mi][i];
      const float sg = xl / (1.0f + __expf(-xl));   // silu
      actE[(size_t)(mbase + i) * INTER + n] = (bf16_t)(sg * accG[mi][i]);
    }
  }
}

// ---------------- Stage 3: fc2 + routing scale + atomic accumulate ----------------
// grid: (HIDDEN/128, NUM_EXPERTS, 4 k-splits), block 256 (8 waves: 2 in M x 4 in N)
__global__ __launch_bounds__(256) void moe_fc2_kernel(
    const bf16_t* __restrict__ act, const float* __restrict__ W2,
    const float* __restrict__ coef, float* __restrict__ out) {
  __shared__ bf16_t ldsA0[TOKENS * LDS_STRIDE];
  __shared__ bf16_t ldsA1[TOKENS * LDS_STRIDE];
  __shared__ bf16_t ldsW [128 * LDS_STRIDE];
  __shared__ float  coefLds[TOKENS];

  const int n0   = blockIdx.x * 128;          // H tile
  const int e    = blockIdx.y;
  const int kbeg = blockIdx.z * (INTER / 4);  // K split
  const int tid  = threadIdx.x;
  const int lane = tid & 31;
  const int wave = tid >> 5;
  const int wm   = wave & 1;
  const int wn   = wave >> 1;

  if (tid < TOKENS) coefLds[tid] = coef[tid * NUM_EXPERTS + e];

  const bf16_t* actE = act + (size_t)e * TOKENS * INTER;
  const float*  w2e  = W2 + (size_t)e * HIDDEN * INTER;

  v8f acc[4][2] = {};
  v4f wr[8];

  // ---- preload stage 0 ----
  #pragma unroll
  for (int i = 0; i < ASYNC_CHUNKS; ++i) {
    const int idx = tid + i * 256;
    const int row = idx >> 3;
    const int c8  = (idx & 7) << 3;
    async_copy_b128(actE + (size_t)row * INTER + kbeg + c8, ldsA0 + row * LDS_STRIDE + c8);
  }
  #pragma unroll
  for (int i = 0; i < 8; ++i) {
    const int idx4 = tid + i * 256;             // 2048 quads per 128x64 fp32 tile
    const int row  = idx4 >> 4;
    const int c4   = (idx4 & 15) << 2;
    wr[i] = __builtin_nontemporal_load((const v4f*)(w2e + (size_t)(n0 + row) * INTER + kbeg + c4));
  }

  const int S = (INTER / 4) / LDS_K;            // 16 stages
  for (int s = 0; s < S; ++s) {
    const bf16_t* ldsAc = (s & 1) ? ldsA1 : ldsA0;
    bf16_t*       ldsAn = (s & 1) ? ldsA0 : ldsA1;

    __syncthreads();

    #pragma unroll
    for (int i = 0; i < 8; ++i) {
      const int idx4 = tid + i * 256;
      const int row  = idx4 >> 4;
      const int c4   = (idx4 & 15) << 2;
      *(v4bf*)(ldsW + row * LDS_STRIDE + c4) = cvt4(wr[i]);
    }

    if (s + 1 < S) {
      const int k1 = kbeg + (s + 1) * LDS_K;
      #pragma unroll
      for (int i = 0; i < ASYNC_CHUNKS; ++i) {
        const int idx = tid + i * 256;
        const int row = idx >> 3;
        const int c8  = (idx & 7) << 3;
        async_copy_b128(actE + (size_t)row * INTER + k1 + c8, ldsAn + row * LDS_STRIDE + c8);
      }
      #pragma unroll
      for (int i = 0; i < 8; ++i) {
        const int idx4 = tid + i * 256;
        const int row  = idx4 >> 4;
        const int c4   = (idx4 & 15) << 2;
        wr[i] = __builtin_nontemporal_load((const v4f*)(w2e + (size_t)(n0 + row) * INTER + k1 + c4));
      }
      wait_asynccnt<ASYNC_CHUNKS>();
    } else {
      wait_asynccnt<0>();
    }
    __syncthreads();

    #pragma unroll
    for (int kk = 0; kk < LDS_K; kk += 32) {
      const int arow = wm * 64 + (lane & 15);
      const int kbA  = kk + ((lane >> 4) << 3);
      v16bf a[4];
      #pragma unroll
      for (int mi = 0; mi < 4; ++mi) a[mi] = load_frag_a(ldsAc, arow + mi * 16, kbA);
      const int bcol = wn * 32 + (lane & 15);
      const int kbB  = kk + ((lane >> 4) << 3);
      #pragma unroll
      for (int nj = 0; nj < 2; ++nj) {
        v16bf b = load_frag_b(ldsW, bcol + nj * 16, kbB);
        #pragma unroll
        for (int mi = 0; mi < 4; ++mi) acc[mi][nj] = wmma_bf16(a[mi], b, acc[mi][nj]);
      }
    }
  }

  #pragma unroll
  for (int mi = 0; mi < 4; ++mi) {
    const int mbase = wm * 64 + mi * 16 + ((lane >> 4) << 3);
    #pragma unroll
    for (int nj = 0; nj < 2; ++nj) {
      const int n = n0 + wn * 32 + nj * 16 + (lane & 15);
      #pragma unroll
      for (int i = 0; i < 8; ++i) {
        const int t = mbase + i;
        atomicAdd(out + (size_t)t * HIDDEN + n, acc[mi][nj][i] * coefLds[t]);
      }
    }
  }
}

extern "C" void kernel_launch(void* const* d_in, const int* in_sizes, int n_in,
                              void* d_out, int out_size, void* d_ws, size_t ws_size,
                              hipStream_t stream) {
  const float* X  = (const float*)d_in[0];   // [128,1024]
  const float* R  = (const float*)d_in[1];   // [128,8]
  const float* W1 = (const float*)d_in[2];   // [8,8192,1024]
  const float* W2 = (const float*)d_in[3];   // [8,1024,4096]
  float* out = (float*)d_out;

  float*  coef = (float*)d_ws;                                  // 4 KB
  bf16_t* Xb   = (bf16_t*)((char*)d_ws + 8192);                 // 256 KB
  bf16_t* act  = (bf16_t*)((char*)d_ws + 8192 + 262144);        // 8 MB

  (void)hipMemsetAsync(d_out, 0, (size_t)TOKENS * HIDDEN * sizeof(float), stream);
  moe_cvt_x_kernel<<<TOKENS * HIDDEN / (256 * 4), 256, 0, stream>>>(X, Xb);
  moe_routing_kernel<<<1, TOKENS, 0, stream>>>(R, coef);
  moe_fc1_kernel<<<dim3(INTER / 64, NUM_EXPERTS), 256, 0, stream>>>(Xb, W1, act);
  moe_fc2_kernel<<<dim3(HIDDEN / 128, NUM_EXPERTS, 4), 256, 0, stream>>>(act, W2, coef, out);
}
